// Encoder_2869038154329
// MI455X (gfx1250) — matrix operand order repaired and checked by reading, not compiled, not run on hardware
//
#include <hip/hip_runtime.h>

// ---------------------------------------------------------------------------
// Types
// ---------------------------------------------------------------------------
typedef __attribute__((ext_vector_type(16))) __bf16 v16bf;
typedef __attribute__((ext_vector_type(8)))  __bf16 v8bf;
typedef __attribute__((ext_vector_type(8)))  float  v8f;
typedef unsigned short     u16;
typedef unsigned int       u32;
typedef unsigned long long u64;

// Model constants
#define S_LEN   2048
#define D_MODEL 1024
#define FFN_H   4096
#define N_HEADS 16
#define HEAD_D  64
#define N_LAYER 6
#define BATCH   4
#define MROWS   (BATCH * S_LEN)   // 8192

__device__ __forceinline__ u16 f2bf(float f) {
    u32 u = __builtin_bit_cast(u32, f);
    u32 r = u + 0x7FFFu + ((u >> 16) & 1u);   // round-to-nearest-even
    return (u16)(r >> 16);
}

__device__ __forceinline__ float lane_bcast(float v, int srcLane) {
    int i = __builtin_amdgcn_ds_bpermute(srcLane << 2, __builtin_bit_cast(int, v));
    return __builtin_bit_cast(float, i);
}

// Async DMA: 16B per lane, global -> LDS, tracked by ASYNCcnt (gfx1250).
__device__ __forceinline__ void async_b128(u32 lds_off, const void* src) {
    asm volatile("global_load_async_to_lds_b128 %0, %1, off"
                 :: "v"(lds_off), "v"((u64)(size_t)src)
                 : "memory");
}
__device__ __forceinline__ void wait_async0() {
#if __has_builtin(__builtin_amdgcn_s_wait_asynccnt)
    __builtin_amdgcn_s_wait_asynccnt(0);
#else
    asm volatile("s_wait_asynccnt 0x0" ::: "memory");
#endif
}

// ---------------------------------------------------------------------------
// f32 -> bf16 conversion (plain, for activations)
// ---------------------------------------------------------------------------
__global__ __launch_bounds__(256) void cvt_bf16_kernel(
    const float* __restrict__ src, u16* __restrict__ dst, int n) {
    int i = blockIdx.x * 256 + threadIdx.x;
    if (i < n) dst[i] = f2bf(src[i]);
}

// f32 [K,N] -> bf16 transposed [N,K], 32x32 LDS tile (coalesced both sides).
__global__ __launch_bounds__(256) void cvt_transpose_kernel(
    const float* __restrict__ src, u16* __restrict__ dst, int K, int N) {
    __shared__ float t[32][33];
    const int n0 = blockIdx.x * 32, k0 = blockIdx.y * 32;
    const int tx = threadIdx.x & 31, ty = threadIdx.x >> 5;
#pragma unroll
    for (int i = 0; i < 4; ++i) {
        int k = ty + i * 8;
        t[k][tx] = src[(size_t)(k0 + k) * N + n0 + tx];
    }
    __syncthreads();
#pragma unroll
    for (int i = 0; i < 4; ++i) {
        int n = ty + i * 8;
        dst[(size_t)(n0 + n) * K + k0 + tx] = f2bf(t[tx][n]);
    }
}

// copy x into residual stream (f32) and produce bf16 operand
__global__ __launch_bounds__(256) void init_kernel(
    const float* __restrict__ x, float* __restrict__ r, u16* __restrict__ xb, int n) {
    int i = blockIdx.x * 256 + threadIdx.x;
    if (i < n) { float v = x[i]; r[i] = v; xb[i] = f2bf(v); }
}

// ---------------------------------------------------------------------------
// Tiled bf16 WMMA GEMM:  C[M,N] = A[M,K] * Bt[N,K]^T + bias
// A row-major bf16, B pre-transposed bf16.  Block tile 256x128x32,
// 256 threads = 8 waves (4x2), wave tile 64x64 (4x4 WMMA accumulators).
// Double-buffered LDS fed by global_load_async_to_lds_b128 (ASYNCcnt).
// ---------------------------------------------------------------------------
#define BM  256
#define BN  128
#define BK  32
#define LDA 40   // sA row stride in halves (pad keeps 16B chunk alignment)
#define LDB 48   // sB row stride in halves (pad keeps 32B chunk alignment)

template <int RELU, int OUT_BF16>
__global__ __launch_bounds__(256) void gemm_bf16_kernel(
    const u16* __restrict__ A, const u16* __restrict__ Bt,
    const float* __restrict__ bias,
    float* __restrict__ Cf, u16* __restrict__ Cb,
    int M, int N, int K) {
    __shared__ alignas(32) u16 sA[2][BM * LDA];
    __shared__ alignas(32) u16 sB[2][BN * LDB];

    const int tid  = threadIdx.x;
    const int lane = tid & 31;
    const int wave = tid >> 5;
    const int wm   = (wave >> 1) * 64;   // wave M offset: 0,64,128,192
    const int wn   = (wave & 1) * 64;    // wave N offset: 0,64
    const int m0   = blockIdx.y * BM;
    const int n0   = blockIdx.x * BN;

    const int koffA = (lane < 16) ? 0 : 16;    // A 16x32 bf16 layout
    const int koffB = (lane < 16) ? 0 : 16;    // B 32x16 bf16 layout
    const int rsel  = (lane & 16) ? 8 : 0;     // C row select per lane group

    v8f acc[4][4] = {};

    auto stage = [&](int bufi, int k0) {
        u16* ab = sA[bufi];
        u16* bb = sB[bufi];
        for (int i = tid; i < BM * 4; i += 256) {          // 4 chunks/thread
            int r = i >> 2, sg = (i & 3) * 8;
            async_b128((u32)(size_t)(ab + r * LDA + sg),
                       A + (size_t)(m0 + r) * K + k0 + sg);
        }
        for (int i = tid; i < BN * 4; i += 256) {          // 2 chunks/thread
            int r = i >> 2, sg = (i & 3) * 8;
            async_b128((u32)(size_t)(bb + r * LDB + sg),
                       Bt + (size_t)(n0 + r) * K + k0 + sg);
        }
    };

    const int nk = K / BK;
    stage(0, 0);
    int bufi = 0;

    for (int kt = 0; kt < nk; ++kt) {
        wait_async0();
        __syncthreads();                       // tile `bufi` resident
        if (kt + 1 < nk) stage(bufi ^ 1, (kt + 1) * BK);
        if (kt + 2 < nk)                       // L2 prefetch of tile after next
            __builtin_prefetch(A + (size_t)(m0 + tid) * K + (kt + 2) * BK, 0, 0);

        const u16* ab = sA[bufi];
        const u16* bb = sB[bufi];
        union AV { v16bf v; v8bf h[2]; };
        AV av[4];
#pragma unroll
        for (int mt = 0; mt < 4; ++mt) {
            int row = wm + mt * 16 + (lane & 15);
            av[mt].h[0] = *(const v8bf*)(ab + row * LDA + koffA);
            av[mt].h[1] = *(const v8bf*)(ab + row * LDA + koffA + 8);
        }
        v16bf bv[4];
#pragma unroll
        for (int nt = 0; nt < 4; ++nt) {
            int col = wn + nt * 16 + (lane & 15);
            bv[nt] = *(const v16bf*)(bb + col * LDB + koffB);
        }
#pragma unroll
        for (int mt = 0; mt < 4; ++mt)
#pragma unroll
            for (int nt = 0; nt < 4; ++nt)
                acc[mt][nt] = __builtin_amdgcn_wmma_f32_16x16x32_bf16(
                    false, av[mt].v, false, bv[nt], (short)0, acc[mt][nt], false, false);
        __syncthreads();                       // all reads of `bufi` done
        bufi ^= 1;
    }

    // ---- epilogue: bias (+ReLU), write f32 or bf16 ----
#pragma unroll
    for (int mt = 0; mt < 4; ++mt) {
#pragma unroll
        for (int nt = 0; nt < 4; ++nt) {
            int col = n0 + wn + nt * 16 + (lane & 15);
            float bb = bias[col];
#pragma unroll
            for (int r = 0; r < 8; ++r) {
                int row = m0 + wm + mt * 16 + r + rsel;
                float v = acc[mt][nt][r] + bb;
                if (RELU) v = fmaxf(v, 0.0f);
                if (OUT_BF16) Cb[(size_t)row * N + col] = f2bf(v);
                else          Cf[(size_t)row * N + col] = v;
            }
        }
    }
}

// ---------------------------------------------------------------------------
// Flash-style attention. qkv layout: [B*S, 3072], head h at cols h*192
// (q: +0, k: +64, v: +128).  Grid: (S/128, H, B), block = 128 (4 waves).
// Each wave owns 32 query rows, iterates keys in tiles of 64 with online
// softmax; WMMA bf16 for QK^T and PV.  K tile staged via async DMA.
// ---------------------------------------------------------------------------
__global__ __launch_bounds__(128) void attn_kernel(
    const u16* __restrict__ qkv, const float* __restrict__ mask,
    u16* __restrict__ vals) {
    __shared__ alignas(32) u16  sK [64 * 64];        // K rows (Bt for QK^T)
    __shared__ alignas(32) u16  sVt[64 * 64];        // V transposed [d][n]
    __shared__ alignas(32) float sS[4][32 * 64];     // per-wave score tile
    __shared__ alignas(32) u16  sP[4][32 * 64];      // per-wave prob tile bf16

    const int tid  = threadIdx.x;
    const int lane = tid & 31;
    const int wave = tid >> 5;
    const int b    = blockIdx.z;
    const int h    = blockIdx.y;
    const int q0   = blockIdx.x * 128 + wave * 32;
    const int QLD  = 3 * D_MODEL;
    const int hq   = h * 192;

    const int koffA = (lane < 16) ? 0 : 16;
    const int koffB = (lane < 16) ? 0 : 16;
    const int rsel  = (lane & 16) ? 8 : 0;

    union AV { v16bf v; v8bf h2[2]; };

    // Q fragments (32 rows x 64 d) held in registers across the key loop
    AV aq[2][2];
#pragma unroll
    for (int mt = 0; mt < 2; ++mt)
#pragma unroll
        for (int ks = 0; ks < 2; ++ks) {
            int row = q0 + mt * 16 + (lane & 15);
            const u16* p = qkv + (size_t)(b * S_LEN + row) * QLD + hq + ks * 32;
            aq[mt][ks].h2[0] = *(const v8bf*)(p + koffA);
            aq[mt][ks].h2[1] = *(const v8bf*)(p + koffA + 8);
        }

    v8f   oacc[2][4] = {};
    float m_run = -1e30f, l_run = 0.0f;
    const int qrow = q0 + lane;   // softmax row owned by this lane

    for (int kt = 0; kt < S_LEN / 64; ++kt) {
        const int k0 = kt * 64;
        // ---- K tile: contiguous rows -> async DMA to LDS ----
        for (int i = tid; i < 64 * 8; i += 128) {
            int r = i >> 3, sg = (i & 7) * 8;
            async_b128((u32)(size_t)(sK + r * 64 + sg),
                       qkv + (size_t)(b * S_LEN + k0 + r) * QLD + hq + 64 + sg);
        }
        // ---- V tile: transpose scatter through VGPRs ----
        for (int i = tid; i < 64 * 8; i += 128) {
            int r = i >> 3, sg = (i & 7) * 8;
            uint4 v = *(const uint4*)(qkv + (size_t)(b * S_LEN + k0 + r) * QLD + hq + 128 + sg);
            u16 tmp[8]; *(uint4*)tmp = v;
#pragma unroll
            for (int j = 0; j < 8; ++j) sVt[(sg + j) * 64 + r] = tmp[j];
        }
        wait_async0();
        __syncthreads();

        // ---- S = Q * K^T  (32x64, k=d=64) ----
        v8f sacc[2][4] = {};
#pragma unroll
        for (int ks = 0; ks < 2; ++ks) {
            v16bf bk[4];
#pragma unroll
            for (int nt = 0; nt < 4; ++nt) {
                int col = nt * 16 + (lane & 15);
                bk[nt] = *(const v16bf*)(sK + col * 64 + ks * 32 + koffB);
            }
#pragma unroll
            for (int mt = 0; mt < 2; ++mt)
#pragma unroll
                for (int nt = 0; nt < 4; ++nt)
                    sacc[mt][nt] = __builtin_amdgcn_wmma_f32_16x16x32_bf16(
                        false, aq[mt][ks].v, false, bk[nt], (short)0, sacc[mt][nt], false, false);
        }

        // ---- dump scaled scores to per-wave LDS tile ----
        float* ss = sS[wave];
#pragma unroll
        for (int mt = 0; mt < 2; ++mt)
#pragma unroll
            for (int nt = 0; nt < 4; ++nt)
#pragma unroll
                for (int r = 0; r < 8; ++r)
                    ss[(mt * 16 + r + rsel) * 64 + nt * 16 + (lane & 15)] =
                        sacc[mt][nt][r] * 0.125f;   // 1/sqrt(64)

        // ---- online softmax: lane owns row `lane` of this wave's 32 rows ----
        const float* mrow = mask + (size_t)qrow * S_LEN + k0;
        float tmax = -1e30f;
        for (int j = 0; j < 64; ++j) {
            float s = ss[lane * 64 + j] + mrow[j];
            ss[lane * 64 + j] = s;
            tmax = fmaxf(tmax, s);
        }
        float newm  = fmaxf(m_run, tmax);
        float alpha = __expf(m_run - newm);
        float lsum  = 0.0f;
        u16* pp = sP[wave];
        for (int j = 0; j < 64; ++j) {
            float e = __expf(ss[lane * 64 + j] - newm);
            lsum += e;
            pp[lane * 64 + j] = f2bf(e);
        }
        l_run = l_run * alpha + lsum;
        m_run = newm;

        // ---- rescale O accumulators by alpha of their rows ----
#pragma unroll
        for (int mt = 0; mt < 2; ++mt)
#pragma unroll
            for (int r = 0; r < 8; ++r) {
                float a = lane_bcast(alpha, mt * 16 + r + rsel);
#pragma unroll
                for (int nt = 0; nt < 4; ++nt) oacc[mt][nt][r] *= a;
            }

        // ---- O += P * V  (32x64, k=keys=64) ----
#pragma unroll
        for (int ks = 0; ks < 2; ++ks) {
            AV ap[2];
#pragma unroll
            for (int mt = 0; mt < 2; ++mt) {
                const u16* base = pp + (mt * 16 + (lane & 15)) * 64 + ks * 32;
                ap[mt].h2[0] = *(const v8bf*)(base + koffA);
                ap[mt].h2[1] = *(const v8bf*)(base + koffA + 8);
            }
            v16bf bvv[4];
#pragma unroll
            for (int nt = 0; nt < 4; ++nt) {
                int dcol = nt * 16 + (lane & 15);
                bvv[nt] = *(const v16bf*)(sVt + dcol * 64 + ks * 32 + koffB);
            }
#pragma unroll
            for (int mt = 0; mt < 2; ++mt)
#pragma unroll
                for (int nt = 0; nt < 4; ++nt)
                    oacc[mt][nt] = __builtin_amdgcn_wmma_f32_16x16x32_bf16(
                        false, ap[mt].v, false, bvv[nt], (short)0, oacc[mt][nt], false, false);
        }
        __syncthreads();   // sK/sVt reused next iteration
    }

    // ---- finalize: divide by l, write vals[b*S+q, h*64+d] as bf16 ----
    float inv = 1.0f / l_run;
#pragma unroll
    for (int mt = 0; mt < 2; ++mt)
#pragma unroll
        for (int r = 0; r < 8; ++r) {
            int arow = mt * 16 + r + rsel;
            float iv = lane_bcast(inv, arow);
#pragma unroll
            for (int nt = 0; nt < 4; ++nt) {
                int dcol = nt * 16 + (lane & 15);
                vals[(size_t)(b * S_LEN + q0 + arow) * D_MODEL + h * HEAD_D + dcol] =
                    f2bf(oacc[mt][nt][r] * iv);
            }
        }
}

// ---------------------------------------------------------------------------
// Residual add + LayerNorm: one block per row of 1024.
// Writes f32 residual stream and bf16 operand for next GEMM.
// ---------------------------------------------------------------------------
__global__ __launch_bounds__(256) void add_ln_kernel(
    const float* __restrict__ a, const float* __restrict__ resid_in,
    const float* __restrict__ gamma, const float* __restrict__ beta,
    float* __restrict__ resid_out, u16* __restrict__ yb) {
    __shared__ float red[256];
    const int tid = threadIdx.x;
    const size_t base = (size_t)blockIdx.x * D_MODEL;

    float v[4]; float s = 0.0f;
#pragma unroll
    for (int i = 0; i < 4; ++i) {
        int c = tid + i * 256;
        v[i] = a[base + c] + resid_in[base + c];
        s += v[i];
    }
    red[tid] = s; __syncthreads();
    for (int st = 128; st > 0; st >>= 1) {
        if (tid < st) red[tid] += red[tid + st];
        __syncthreads();
    }
    float mean = red[0] * (1.0f / D_MODEL);
    __syncthreads();

    float vs = 0.0f;
#pragma unroll
    for (int i = 0; i < 4; ++i) { float d = v[i] - mean; vs += d * d; }
    red[tid] = vs; __syncthreads();
    for (int st = 128; st > 0; st >>= 1) {
        if (tid < st) red[tid] += red[tid + st];
        __syncthreads();
    }
    float rstd = rsqrtf(red[0] * (1.0f / D_MODEL) + 1e-5f);

#pragma unroll
    for (int i = 0; i < 4; ++i) {
        int c = tid + i * 256;
        float t = (v[i] - mean) * rstd * gamma[c] + beta[c];
        resid_out[base + c] = t;
        yb[base + c]        = f2bf(t);
    }
}

// ---------------------------------------------------------------------------
// Host orchestration
// ---------------------------------------------------------------------------
extern "C" void kernel_launch(void* const* d_in, const int* in_sizes, int n_in,
                              void* d_out, int out_size, void* d_ws, size_t ws_size,
                              hipStream_t stream) {
    (void)in_sizes; (void)n_in; (void)out_size; (void)ws_size;

    const float* x    = (const float*)d_in[0];
    const float* mask = (const float*)d_in[1];
    const float* Wqkv = (const float*)d_in[2];
    const float* bqkv = (const float*)d_in[3];
    const float* Wo   = (const float*)d_in[4];
    const float* bo   = (const float*)d_in[5];
    const float* g1   = (const float*)d_in[6];
    const float* be1  = (const float*)d_in[7];
    const float* W1   = (const float*)d_in[8];
    const float* b1   = (const float*)d_in[9];
    const float* W2   = (const float*)d_in[10];
    const float* b2   = (const float*)d_in[11];
    const float* g2   = (const float*)d_in[12];
    const float* be2  = (const float*)d_in[13];

    char* ws = (char*)d_ws;
    size_t off = 0;
    auto alloc = [&](size_t bytes) -> void* {
        void* p = ws + off;
        off += (bytes + 255) & ~(size_t)255;
        return p;
    };

    float* r_cur  = (float*)alloc((size_t)MROWS * D_MODEL * 4);
    float* r_nxt  = (float*)alloc((size_t)MROWS * D_MODEL * 4);
    float* t0     = (float*)alloc((size_t)MROWS * D_MODEL * 4);
    u16*   xb     = (u16*)  alloc((size_t)MROWS * D_MODEL * 2);
    u16*   y1b    = (u16*)  alloc((size_t)MROWS * D_MODEL * 2);
    u16*   qkv_b  = (u16*)  alloc((size_t)MROWS * 3 * D_MODEL * 2);
    u16*   vals_b = (u16*)  alloc((size_t)MROWS * D_MODEL * 2);
    u16*   h_b    = (u16*)  alloc((size_t)MROWS * FFN_H * 2);
    u16*   wqkv_t = (u16*)  alloc((size_t)D_MODEL * 3 * D_MODEL * 2);   // [3072,1024]
    u16*   wo_t   = (u16*)  alloc((size_t)D_MODEL * D_MODEL * 2);       // [1024,1024]
    u16*   w1_t   = (u16*)  alloc((size_t)D_MODEL * FFN_H * 2);         // [4096,1024]
    u16*   w2_t   = (u16*)  alloc((size_t)FFN_H * D_MODEL * 2);         // [1024,4096]

    const int nAct = MROWS * D_MODEL;
    init_kernel<<<nAct / 256, 256, 0, stream>>>(x, r_cur, xb, nAct);

    for (int l = 0; l < N_LAYER; ++l) {
        // per-layer weight conversion to bf16, transposed to [N,K]
        size_t nqkv = (size_t)D_MODEL * 3 * D_MODEL;
        size_t nd2  = (size_t)D_MODEL * D_MODEL;
        size_t ndf  = (size_t)D_MODEL * FFN_H;
        cvt_transpose_kernel<<<dim3(3 * D_MODEL / 32, D_MODEL / 32), 256, 0, stream>>>(
            Wqkv + l * nqkv, wqkv_t, D_MODEL, 3 * D_MODEL);
        cvt_transpose_kernel<<<dim3(D_MODEL / 32, D_MODEL / 32), 256, 0, stream>>>(
            Wo + l * nd2, wo_t, D_MODEL, D_MODEL);
        cvt_transpose_kernel<<<dim3(FFN_H / 32, D_MODEL / 32), 256, 0, stream>>>(
            W1 + l * ndf, w1_t, D_MODEL, FFN_H);
        cvt_transpose_kernel<<<dim3(D_MODEL / 32, FFN_H / 32), 256, 0, stream>>>(
            W2 + l * ndf, w2_t, FFN_H, D_MODEL);

        // QKV projection: [8192,1024] x [1024,3072] -> bf16
        {
            dim3 grid(3 * D_MODEL / BN, MROWS / BM);
            gemm_bf16_kernel<0, 1><<<grid, 256, 0, stream>>>(
                xb, wqkv_t, bqkv + (size_t)l * 3 * D_MODEL,
                nullptr, qkv_b, MROWS, 3 * D_MODEL, D_MODEL);
        }
        // attention
        {
            dim3 grid(S_LEN / 128, N_HEADS, BATCH);
            attn_kernel<<<grid, 128, 0, stream>>>(qkv_b, mask, vals_b);
        }
        // output projection -> f32
        {
            dim3 grid(D_MODEL / BN, MROWS / BM);
            gemm_bf16_kernel<0, 0><<<grid, 256, 0, stream>>>(
                vals_b, wo_t, bo + (size_t)l * D_MODEL,
                t0, nullptr, MROWS, D_MODEL, D_MODEL);
        }
        // residual + LN1
        add_ln_kernel<<<MROWS, 256, 0, stream>>>(
            t0, r_cur, g1 + (size_t)l * D_MODEL, be1 + (size_t)l * D_MODEL, r_nxt, y1b);
        // FFN1 + ReLU -> bf16
        {
            dim3 grid(FFN_H / BN, MROWS / BM);
            gemm_bf16_kernel<1, 1><<<grid, 256, 0, stream>>>(
                y1b, w1_t, b1 + (size_t)l * FFN_H,
                nullptr, h_b, MROWS, FFN_H, D_MODEL);
        }
        // FFN2 -> f32
        {
            dim3 grid(D_MODEL / BN, MROWS / BM);
            gemm_bf16_kernel<0, 0><<<grid, 256, 0, stream>>>(
                h_b, w2_t, b2 + (size_t)l * D_MODEL,
                t0, nullptr, MROWS, D_MODEL, FFN_H);
        }
        // residual + LN2 -> next layer input
        add_ln_kernel<<<MROWS, 256, 0, stream>>>(
            t0, r_nxt, g2 + (size_t)l * D_MODEL, be2 + (size_t)l * D_MODEL, r_cur, xb);
    }

    hipMemcpyAsync(d_out, r_cur, (size_t)MROWS * D_MODEL * sizeof(float),
                   hipMemcpyDeviceToDevice, stream);
}
